// I_CRGCN_57002805952693
// MI455X (gfx1250) — compile-verified, matrix operand order checked by reading, not run on hardware
//
#include <hip/hip_runtime.h>
#include <hip/hip_bf16.h>

#define N_USERS_P1 50001
#define N_ITEMS_P1 100001
#define NTOT (N_USERS_P1 + N_ITEMS_P1)   /* 150002 */
#define D 64
#define N_BEH 2
#define N_EDGES 2000000
#define BATCH 2048
#define REG_WEIGHT 1.0e-4f
#define EPSV 1.0e-9f
#define GAMMAV 1.0e-10f

typedef __attribute__((ext_vector_type(16))) _Float16 v16h;
typedef __attribute__((ext_vector_type(8)))  float    v8f;

__device__ __forceinline__ float waveReduceSum(float v) {
#pragma unroll
    for (int off = 16; off > 0; off >>= 1)
        v += __shfl_xor(v, off, 32);
    return v;
}

__global__ void k_init_red(float* red) {
    if (threadIdx.x < 16) red[threadIdx.x] = 0.f;
}

// total = concat(user_emb, item_emb), float4-vectorized
__global__ void k_concat(const float4* __restrict__ ue, const float4* __restrict__ ie,
                         float4* __restrict__ total) {
    int i = blockIdx.x * blockDim.x + threadIdx.x;
    const int nu4 = N_USERS_P1 * D / 4;
    const int nt4 = NTOT * D / 4;
    if (i < nt4) total[i] = (i < nu4) ? ue[i] : ie[i - nu4];
}

// normB[b*N+n] = 1 / (sqrt(relu(old*denom_w) + now) + eps)
__global__ void k_norm(const float* __restrict__ oldd, const float* __restrict__ nowd,
                       const float* __restrict__ denw, float* __restrict__ normB) {
    int i = blockIdx.x * blockDim.x + threadIdx.x;
    if (i < N_BEH * NTOT) {
        float od  = oldd[i] * denw[0];
        float den = sqrtf(fmaxf(od, 0.f) + nowd[i]);
        normB[i] = 1.f / (den + EPSV);
    }
}

__global__ void k_zero(float4* __restrict__ acc) {
    int i = blockIdx.x * blockDim.x + threadIdx.x;
    if (i < NTOT * D / 4) {
        float4 z; z.x = z.y = z.z = z.w = 0.f;
        acc[i] = z;
    }
}

// acc[dst] += norm[src] * total[src]  (one thread per (edge, float4-chunk))
__global__ void k_segsum(const int* __restrict__ src, const int* __restrict__ dst,
                         const float* __restrict__ normB, const float* __restrict__ total,
                         float* __restrict__ acc, int b) {
    long long tid = (long long)blockIdx.x * blockDim.x + threadIdx.x;
    if (tid >= (long long)N_EDGES * 16) return;
    int e = (int)(tid >> 4), q = (int)(tid & 15);
    int s = src[(size_t)b * N_EDGES + e];
    int d = dst[(size_t)b * N_EDGES + e];
    // gfx1250 global_prefetch of a future edge's source row (hide gather latency)
    int ef = e + 8192;
    if (ef < N_EDGES) {
        int sf = src[(size_t)b * N_EDGES + ef];
        __builtin_prefetch(&total[(size_t)sf * D + q * 4], 0, 1);
    }
    float ns = normB[(size_t)b * NTOT + s];
    const float4 v = ((const float4*)(total + (size_t)s * D))[q];
    float* ap = acc + (size_t)d * D + (size_t)q * 4;
    atomicAdd(ap + 0, v.x * ns);
    atomicAdd(ap + 1, v.y * ns);
    atomicAdd(ap + 2, v.z * ns);
    atomicAdd(ap + 3, v.w * ns);
}

// mean of [total, cw00*old0*rscale+cw01*total, cw10*old1*rscale+cw11*(norm*acc)],
// L2-row-normalize, residual add into total. One wave32 per node row.
__global__ void k_combine(const float* __restrict__ oldd, const float* __restrict__ nowd,
                          const float* __restrict__ denw, const float* __restrict__ oldw,
                          const float* __restrict__ convw, const float* __restrict__ last_st,
                          const float* __restrict__ acc, float* __restrict__ total, int b) {
    int lane = threadIdx.x & 31;
    int wid  = threadIdx.x >> 5;
    int n = blockIdx.x * 8 + wid;
    if (n >= NTOT) return;
    float od  = oldd[(size_t)b * NTOT + n];
    float nd  = nowd[(size_t)b * NTOT + n];
    float den = sqrtf(fmaxf(od * denw[0], 0.f) + nd);
    float nrm = 1.f / (den + EPSV);
    float rsc = sqrtf(fmaxf(od * oldw[0], 0.f)) / (den + EPSV);
    float cw00 = convw[b * 4 + 0], cw01 = convw[b * 4 + 1];
    float cw10 = convw[b * 4 + 2], cw11 = convw[b * 4 + 3];
    size_t base = (size_t)n * D + (size_t)lane * 2;
    float2 t  = *(const float2*)(total + base);
    float2 a  = *(const float2*)(acc + base);
    float2 o0 = *(const float2*)(last_st + ((size_t)(b * 2 + 0) * NTOT + n) * D + lane * 2);
    float2 o1 = *(const float2*)(last_st + ((size_t)(b * 2 + 1) * NTOT + n) * D + lane * 2);
    float mx = (t.x + (cw00 * o0.x * rsc + cw01 * t.x) + (cw10 * o1.x * rsc + cw11 * (nrm * a.x))) * (1.f / 3.f);
    float my = (t.y + (cw00 * o0.y * rsc + cw01 * t.y) + (cw10 * o1.y * rsc + cw11 * (nrm * a.y))) * (1.f / 3.f);
    float ssq = waveReduceSum(mx * mx + my * my);
    float inv = 1.f / fmaxf(sqrtf(ssq), 1e-12f);
    float2 outv;
    outv.x = mx * inv + t.x;
    outv.y = my * inv + t.y;
    *(float2*)(total + base) = outv;
}

// BPR loss via v_wmma_f32_16x16x32_f16. Gathered rows are DMA'd into LDS with
// gfx1250 async-to-LDS engine (global_load_async_to_lds_b128, ASYNCcnt).
__global__ void __launch_bounds__(32) k_bpr_wmma(const float* __restrict__ total,
                                                 const int* __restrict__ batch,
                                                 float* __restrict__ red, int b) {
    __shared__ float Uf[16][D];   // 4 KB each
    __shared__ float Pf[16][D];
    __shared__ float Nf[16][D];
    __shared__ float Dt[16][16];
    __shared__ int   ridx[3][16];
    int lane = threadIdx.x;
    int tile = blockIdx.x;
    if (lane < 16) {
        int m = tile * 16 + lane;
        const int* bd = batch + ((size_t)m * N_BEH + b) * 3;
        ridx[0][lane] = bd[0];               // user row in total
        ridx[1][lane] = N_USERS_P1 + bd[1];  // pos item row
        ridx[2][lane] = N_USERS_P1 + bd[2];  // neg item row
    }
    __syncthreads();
    // Async DMA: 16 rows x 256B per matrix = 256 b128 chunks; 8 per lane, no divergence.
    for (int i = lane; i < 16 * 16; i += 32) {
        int r = i >> 4, q = i & 15;
        const float* gU = total + (size_t)ridx[0][r] * D + q * 4;
        const float* gP = total + (size_t)ridx[1][r] * D + q * 4;
        const float* gN = total + (size_t)ridx[2][r] * D + q * 4;
        unsigned lU = (unsigned)(uintptr_t)&Uf[r][q * 4];
        unsigned lP = (unsigned)(uintptr_t)&Pf[r][q * 4];
        unsigned lN = (unsigned)(uintptr_t)&Nf[r][q * 4];
        asm volatile("global_load_async_to_lds_b128 %0, %1, off" :: "v"(lU), "v"(gU) : "memory");
        asm volatile("global_load_async_to_lds_b128 %0, %1, off" :: "v"(lP), "v"(gP) : "memory");
        asm volatile("global_load_async_to_lds_b128 %0, %1, off" :: "v"(lN), "v"(gN) : "memory");
    }
    asm volatile("s_wait_asynccnt 0" ::: "memory");
    __syncthreads();
    // Build WMMA operands per the 16-bit 16x32 A layout:
    // lane L holds row M=L%16; half h -> K = (h<8 ? h : h+8) + (L>=16 ? 8 : 0)
    int mcol = lane & 15;
    int kofs = (lane >= 16) ? 8 : 0;
    v16h a0, a1, p0, p1, g0, g1;
#pragma unroll
    for (int h = 0; h < 16; ++h) {
        int k = ((h < 8) ? h : h + 8) + kofs;
        a0[h] = (_Float16)Uf[mcol][k];  a1[h] = (_Float16)Uf[mcol][k + 32];
        p0[h] = (_Float16)Pf[mcol][k];  p1[h] = (_Float16)Pf[mcol][k + 32];
        g0[h] = (_Float16)Nf[mcol][k];  g1[h] = (_Float16)Nf[mcol][k + 32];
    }
    int mrow = (lane >= 16) ? 8 : 0;
    float s0 = 0.f, s1 = 0.f;
    // positive scores: D = Uf * Pos^T over K=64 (two chained K=32 WMMAs)
    v8f c = {};
    c = __builtin_amdgcn_wmma_f32_16x16x32_f16(false, a0, false, p0, (short)0, c, false, false);
    c = __builtin_amdgcn_wmma_f32_16x16x32_f16(false, a1, false, p1, (short)0, c, false, false);
#pragma unroll
    for (int r = 0; r < 8; ++r) Dt[mrow + r][mcol] = c[r];
    __syncthreads();
    if (lane < 16) s0 = Dt[lane][lane];
    __syncthreads();
    // negative scores
    v8f c2 = {};
    c2 = __builtin_amdgcn_wmma_f32_16x16x32_f16(false, a0, false, g0, (short)0, c2, false, false);
    c2 = __builtin_amdgcn_wmma_f32_16x16x32_f16(false, a1, false, g1, (short)0, c2, false, false);
#pragma unroll
    for (int r = 0; r < 8; ++r) Dt[mrow + r][mcol] = c2[r];
    __syncthreads();
    float v = 0.f;
    if (lane < 16) {
        s1 = Dt[lane][lane];
        float x   = s0 - s1;
        float sig = 1.f / (1.f + __expf(-x));
        v = -__logf(GAMMAV + sig);
    }
    v = waveReduceSum(v);
    if (lane == 0) atomicAdd(&red[b], v);
}

// sum of squares reduction (Frobenius^2) into red[slot]
__global__ void k_ssq(const float* __restrict__ x, long long n,
                      float* __restrict__ red, int slot) {
    long long i = (long long)blockIdx.x * blockDim.x + threadIdx.x;
    long long stride = (long long)gridDim.x * blockDim.x;
    float s = 0.f;
    for (; i < n; i += stride) { float v = x[i]; s += v * v; }
    s = waveReduceSum(s);
    __shared__ float wsum[8];
    int lane = threadIdx.x & 31, wid = threadIdx.x >> 5;
    if (lane == 0) wsum[wid] = s;
    __syncthreads();
    if (threadIdx.x == 0) {
        float t = 0.f;
#pragma unroll
        for (int w = 0; w < 8; ++w) t += wsum[w];
        atomicAdd(&red[slot], t);
    }
}

__global__ void k_final(const float* __restrict__ red, float* __restrict__ out) {
    if (threadIdx.x == 0 && blockIdx.x == 0) {
        float bpr  = red[0] * (1.f / (float)BATCH) + red[1] * (1.f / (float)BATCH);
        float embl = (sqrtf(red[2]) + sqrtf(red[3])) * (1.f / (float)N_ITEMS_P1);
        out[0] = bpr + REG_WEIGHT * embl;
    }
}

extern "C" void kernel_launch(void* const* d_in, const int* in_sizes, int n_in,
                              void* d_out, int out_size, void* d_ws, size_t ws_size,
                              hipStream_t stream) {
    const float* user_emb = (const float*)d_in[0];
    const float* item_emb = (const float*)d_in[1];
    const float* now_deg  = (const float*)d_in[2];
    const float* old_deg  = (const float*)d_in[3];
    const float* last_st  = (const float*)d_in[4];
    const float* denom_w  = (const float*)d_in[5];
    const float* oldsc_w  = (const float*)d_in[6];
    const float* conv_w   = (const float*)d_in[7];
    const int*   src      = (const int*)d_in[8];
    const int*   dst      = (const int*)d_in[9];
    const int*   batch    = (const int*)d_in[10];
    float* out = (float*)d_out;

    // workspace layout (floats): red[16] | normB[2*N] | total[N*D] | acc[N*D]
    float* ws = (float*)d_ws;
    size_t off = 0;
    float* red    = ws + off; off += 16;
    float* normB  = ws + off; off += (size_t)N_BEH * NTOT;
    off = (off + 15) & ~(size_t)15;
    float* totalB = ws + off; off += (size_t)NTOT * D;
    float* accB   = ws + off; off += (size_t)NTOT * D;
    (void)ws_size; (void)in_sizes; (void)n_in; (void)out_size;

    k_init_red<<<1, 32, 0, stream>>>(red);
    {
        int n4 = NTOT * D / 4;
        k_concat<<<(n4 + 255) / 256, 256, 0, stream>>>(
            (const float4*)user_emb, (const float4*)item_emb, (float4*)totalB);
    }
    k_norm<<<(N_BEH * NTOT + 255) / 256, 256, 0, stream>>>(old_deg, now_deg, denom_w, normB);

    for (int b = 0; b < N_BEH; ++b) {
        int n4 = NTOT * D / 4;
        k_zero<<<(n4 + 255) / 256, 256, 0, stream>>>((float4*)accB);
        long long work = (long long)N_EDGES * 16;
        k_segsum<<<(int)((work + 255) / 256), 256, 0, stream>>>(src, dst, normB, totalB, accB, b);
        k_combine<<<(NTOT + 7) / 8, 256, 0, stream>>>(old_deg, now_deg, denom_w, oldsc_w,
                                                      conv_w, last_st, accB, totalB, b);
        k_bpr_wmma<<<BATCH / 16, 32, 0, stream>>>(totalB, batch, red, b);
    }

    k_ssq<<<1024, 256, 0, stream>>>(user_emb, (long long)N_USERS_P1 * D, red, 2);
    k_ssq<<<1024, 256, 0, stream>>>(item_emb, (long long)N_ITEMS_P1 * D, red, 3);
    k_final<<<1, 32, 0, stream>>>(red, out);
}
